// GraphNeuralNetwork_71184787964246
// MI455X (gfx1250) — matrix operand order repaired and checked by reading, not compiled, not run on hardware
//
#include <hip/hip_runtime.h>
#include <hip/hip_bf16.h>

// ---------------------------------------------------------------------------
// GAT GNN forward for MI455X (gfx1250).
// Dense x@W on v_wmma_f32_16x16x32_bf16 (bf16 inputs, fp32 accumulate);
// edge softmax + scatter aggregation in fp32 (bandwidth/L2 bound path).
// ---------------------------------------------------------------------------

typedef __bf16 bf16_t;
typedef __attribute__((ext_vector_type(8)))  __bf16 v8bf;
typedef __attribute__((ext_vector_type(16))) __bf16 v16bf;
typedef __attribute__((ext_vector_type(8)))  float  v8f;

#define N_NODES 50000
#define N_EDGES 400000
#define N_GRAPH 64
#define DIM_IN  128
#define DIM_HID 128
#define N_HEADS 4

// ----------------------------- helpers -------------------------------------

__device__ __forceinline__ unsigned fenc(float x) {
  unsigned u = __float_as_uint(x);
  return (u & 0x80000000u) ? ~u : (u | 0x80000000u);   // order-preserving key
}
__device__ __forceinline__ float fdec(unsigned k) {
  return __uint_as_float((k & 0x80000000u) ? (k & 0x7FFFFFFFu) : ~k);
}

// ------------------------- conversion / prep kernels ------------------------

__global__ void k_f32_to_bf16(const float* __restrict__ src,
                              bf16_t* __restrict__ dst, int n) {
  int i = blockIdx.x * blockDim.x + threadIdx.x;
  if (i < n) dst[i] = (bf16_t)src[i];
}

// W[K, Nout] (row-major) -> Wt[Nout, K] bf16 so B-fragments are K-contiguous.
__global__ void k_w_transpose_bf16(const float* __restrict__ W,
                                   bf16_t* __restrict__ Wt, int K, int Nout) {
  int i = blockIdx.x * blockDim.x + threadIdx.x;
  if (i >= K * Nout) return;
  int k = i / Nout, n = i - k * Nout;
  Wt[(size_t)n * K + k] = (bf16_t)W[i];
}

// ------------------------------ WMMA GEMM ----------------------------------
// C[M, Nout] = A[M, K] (bf16, row-major) * Bt[Nout, K] (bf16, K-contiguous).
// One wave -> 16 x 64 output strip; 8 waves / block. K, Nout multiples of 32/64;
// M multiple of 16 (50000 = 3125 * 16).

__global__ __launch_bounds__(256)
void k_gemm_bf16_wmma(const bf16_t* __restrict__ A, const bf16_t* __restrict__ Bt,
                      float* __restrict__ C, int Mrows, int K, int Nout) {
  const int lane = threadIdx.x & 31;
  const int wave = threadIdx.x >> 5;
  const int colT = Nout >> 6;
  const int rowT = Mrows >> 4;
  const int tiles = colT * rowT;
  int t = blockIdx.x * 8 + wave;
  if (t >= tiles) return;
  const int mt = t / colT;
  const int nt = t - mt * colT;
  const int l  = lane & 15;
  const int hi = lane >> 4;

  v8f acc0 = {}, acc1 = {}, acc2 = {}, acc3 = {};

  // A fragment (16x32 bf16): lane<16 holds row l, K = k0..k0+15;
  // lane>=16 holds row l, K = k0+16..k0+31 (matches B's contiguous order).
  const bf16_t* arow = A + (size_t)(mt * 16 + l) * K + hi * 16;
  // B fragment (32x16): lane<16 holds col l, K = k0..k0+15;
  // lane>=16 holds col l, K = k0+16..k0+31 -> one contiguous 16-run per lane.
  const bf16_t* b0 = Bt + (size_t)(nt * 64 +  0 + l) * K + hi * 16;
  const bf16_t* b1 = Bt + (size_t)(nt * 64 + 16 + l) * K + hi * 16;
  const bf16_t* b2 = Bt + (size_t)(nt * 64 + 32 + l) * K + hi * 16;
  const bf16_t* b3 = Bt + (size_t)(nt * 64 + 48 + l) * K + hi * 16;

  for (int k0 = 0; k0 < K; k0 += 32) {
    v8bf alo = *(const v8bf*)(arow + k0);
    v8bf ahi = *(const v8bf*)(arow + k0 + 8);
    v16bf av = __builtin_shufflevector(alo, ahi, 0,1,2,3,4,5,6,7,
                                                 8,9,10,11,12,13,14,15);
    v16bf vb0 = *(const v16bf*)(b0 + k0);
    v16bf vb1 = *(const v16bf*)(b1 + k0);
    v16bf vb2 = *(const v16bf*)(b2 + k0);
    v16bf vb3 = *(const v16bf*)(b3 + k0);
    acc0 = __builtin_amdgcn_wmma_f32_16x16x32_bf16(false, av, false, vb0, (short)0, acc0, false, false);
    acc1 = __builtin_amdgcn_wmma_f32_16x16x32_bf16(false, av, false, vb1, (short)0, acc1, false, false);
    acc2 = __builtin_amdgcn_wmma_f32_16x16x32_bf16(false, av, false, vb2, (short)0, acc2, false, false);
    acc3 = __builtin_amdgcn_wmma_f32_16x16x32_bf16(false, av, false, vb3, (short)0, acc3, false, false);
  }

  // C/D layout: VGPR r -> M = r (lanes 0-15) / r+8 (lanes 16-31), N = lane&15.
  float* crow = C + (size_t)(mt * 16 + hi * 8) * Nout + nt * 64 + l;
#pragma unroll
  for (int r = 0; r < 8; ++r) {
    crow[(size_t)r * Nout +  0] = acc0[r];
    crow[(size_t)r * Nout + 16] = acc1[r];
    crow[(size_t)r * Nout + 32] = acc2[r];
    crow[(size_t)r * Nout + 48] = acc3[r];
  }
}

// --------------------------- attention kernels ------------------------------

// Per (node, head): a_s = <h, a_src>, a_d = <h, a_dst>; also init segment-max
// key to enc(-inf) and denom to 0 (re-done every call: ws is poisoned once).
__global__ void k_attn_coef(const float* __restrict__ H, const float* __restrict__ a_src,
                            const float* __restrict__ a_dst, float* __restrict__ as_o,
                            float* __restrict__ ad_o, unsigned* __restrict__ mbuf,
                            float* __restrict__ denom, int n, int heads, int d) {
  int i = blockIdx.x * blockDim.x + threadIdx.x;
  if (i >= n * heads) return;
  int node = i / heads, h = i - node * heads;
  const float* row = H + (size_t)node * heads * d + (size_t)h * d;
  float s = 0.f, t = 0.f;
  for (int j = 0; j < d; ++j) {
    float v = row[j];
    s += v * a_src[h * d + j];
    t += v * a_dst[h * d + j];
  }
  as_o[i] = s;
  ad_o[i] = t;
  mbuf[i] = 0x007FFFFFu;   // enc(-inf)
  denom[i] = 0.f;
}

// Pass 1: e = leaky_relu(a_s[src] + a_d[dst], 0.2); segment max via atomicMax.
// Edge ids >= E are the implicit self loops (src = dst = e - E).
__global__ void k_edge_logits(const int* __restrict__ ei, const float* __restrict__ as_,
                              const float* __restrict__ ad_, float* __restrict__ ebuf,
                              unsigned* __restrict__ mbuf, int heads) {
  int i = blockIdx.x * blockDim.x + threadIdx.x;
  int total = (N_EDGES + N_NODES) * heads;
  if (i >= total) return;
  int e = i / heads, h = i - e * heads;
  int s, dst;
  if (e < N_EDGES) { s = ei[e]; dst = ei[N_EDGES + e]; }
  else             { s = dst = e - N_EDGES; }
  float v = as_[s * heads + h] + ad_[dst * heads + h];
  v = v > 0.f ? v : 0.2f * v;
  ebuf[i] = v;
  atomicMax(&mbuf[dst * heads + h], fenc(v));
}

// Pass 2: w = exp(e - max[dst]); segment sum via atomicAdd.
__global__ void k_edge_exp(const int* __restrict__ ei, float* __restrict__ ebuf,
                           const unsigned* __restrict__ mbuf, float* __restrict__ denom,
                           int heads) {
  int i = blockIdx.x * blockDim.x + threadIdx.x;
  int total = (N_EDGES + N_NODES) * heads;
  if (i >= total) return;
  int e = i / heads, h = i - e * heads;
  int dst = (e < N_EDGES) ? ei[N_EDGES + e] : e - N_EDGES;
  float w = __expf(ebuf[i] - fdec(mbuf[dst * heads + h]));
  ebuf[i] = w;
  atomicAdd(&denom[dst * heads + h], w);
}

// Pass 3: out[dst, f] += h[src, f] * w / (denom[dst] + 1e-16). Feature-parallel.
__global__ void k_edge_aggregate(const int* __restrict__ ei, const float* __restrict__ ebuf,
                                 const float* __restrict__ denom, const float* __restrict__ H,
                                 float* __restrict__ out, int heads, int d) {
  long long i = (long long)blockIdx.x * blockDim.x + threadIdx.x;
  int hd = heads * d;
  long long total = (long long)(N_EDGES + N_NODES) * hd;
  if (i >= total) return;
  int e = (int)(i / hd);
  int f = (int)(i - (long long)e * hd);
  int h = f / d;
  int s, dst;
  if (e < N_EDGES) { s = ei[e]; dst = ei[N_EDGES + e]; }
  else             { s = dst = e - N_EDGES; }
  float alpha = ebuf[(size_t)e * heads + h] / (denom[dst * heads + h] + 1e-16f);
  atomicAdd(&out[(size_t)dst * hd + f], H[(size_t)s * hd + f] * alpha);
}

// ------------------- bias + BN (+ReLU) + bf16 convert ------------------------

__global__ void k_bias_bn_act(const float* __restrict__ x, const float* __restrict__ bias,
                              const float* __restrict__ g, const float* __restrict__ bb,
                              const float* __restrict__ mm, const float* __restrict__ vv,
                              float* __restrict__ of32, bf16_t* __restrict__ obf,
                              int n, int f, int do_relu) {
  long long i = (long long)blockIdx.x * blockDim.x + threadIdx.x;
  long long total = (long long)n * f;
  if (i >= total) return;
  int c = (int)(i % f);
  float y = x[i] + bias[c];
  y = (y - mm[c]) * rsqrtf(vv[c] + 1e-5f) * g[c] + bb[c];
  if (do_relu) y = fmaxf(y, 0.f);
  if (of32) of32[i] = y;
  if (obf)  obf[i]  = (bf16_t)y;
}

// ------------------------------ pooling + MLP --------------------------------

__global__ void k_pool(const float* __restrict__ x, const int* __restrict__ batch,
                       float* __restrict__ sums, float* __restrict__ cnt, int n, int f) {
  int i = blockIdx.x * blockDim.x + threadIdx.x;
  if (i >= n * f) return;
  int node = i / f, c = i - node * f;
  int g = batch[node];
  atomicAdd(&sums[g * f + c], x[i]);
  if (c == 0) atomicAdd(&cnt[g], 1.0f);
}

__global__ void k_classifier(const float* __restrict__ sums, const float* __restrict__ cnt,
                             const float* __restrict__ W1, const float* __restrict__ b1,
                             const float* __restrict__ W2, const float* __restrict__ b2,
                             const float* __restrict__ W3, const float* __restrict__ b3,
                             float* __restrict__ out) {
  int g = threadIdx.x;
  if (g >= N_GRAPH) return;
  float inv = 1.0f / fmaxf(cnt[g], 1.0f);
  float h1[64];
  for (int o = 0; o < 64; ++o) {
    float s = b1[o];
    for (int j = 0; j < 128; ++j) s += sums[g * 128 + j] * inv * W1[j * 64 + o];
    h1[o] = fmaxf(s, 0.f);
  }
  float h2[32];
  for (int o = 0; o < 32; ++o) {
    float s = b2[o];
    for (int j = 0; j < 64; ++j) s += h1[j] * W2[j * 32 + o];
    h2[o] = fmaxf(s, 0.f);
  }
  float s = b3[0];
  for (int j = 0; j < 32; ++j) s += h2[j] * W3[j];
  out[g] = 1.0f / (1.0f + __expf(-s));
}

// ------------------------------- launch --------------------------------------

extern "C" void kernel_launch(void* const* d_in, const int* in_sizes, int n_in,
                              void* d_out, int out_size, void* d_ws, size_t ws_size,
                              hipStream_t stream) {
  (void)in_sizes; (void)n_in; (void)out_size; (void)ws_size;

  // ---- input pointers (setup_inputs() insertion order) ----
  const float* x = (const float*)d_in[0];
  struct Layer { const float *W, *as, *ad, *b, *g, *bb, *m, *v; };
  Layer lay[3];
  for (int i = 0; i < 3; ++i) {
    int base = 1 + i * 8;
    lay[i].W  = (const float*)d_in[base + 0];
    lay[i].as = (const float*)d_in[base + 1];
    lay[i].ad = (const float*)d_in[base + 2];
    lay[i].b  = (const float*)d_in[base + 3];
    lay[i].g  = (const float*)d_in[base + 4];
    lay[i].bb = (const float*)d_in[base + 5];
    lay[i].m  = (const float*)d_in[base + 6];
    lay[i].v  = (const float*)d_in[base + 7];
  }
  const float* cW[3]; const float* cB[3];
  for (int i = 0; i < 3; ++i) {
    cW[i] = (const float*)d_in[25 + 2 * i];
    cB[i] = (const float*)d_in[26 + 2 * i];
  }
  const int* edge_index = (const int*)d_in[31];
  const int* batch      = (const int*)d_in[32];
  float* out = (float*)d_out;

  // ---- workspace layout ----
  char* ws = (char*)d_ws;
  const size_t SZ_FEAT = (size_t)N_NODES * 512 * sizeof(float);   // 102.4 MB
  size_t off = 0;
  float*  bufA  = (float*)(ws + off);  off += SZ_FEAT;            // GEMM out / pooled in
  float*  bufB  = (float*)(ws + off);  off += SZ_FEAT;            // aggregation out
  bf16_t* XB    = (bf16_t*)(ws + off); off += (size_t)N_NODES * 512 * 2;
  bf16_t* WT0   = (bf16_t*)(ws + off); off += (size_t)512 * 128 * 2;
  bf16_t* WT1   = (bf16_t*)(ws + off); off += (size_t)512 * 512 * 2;
  bf16_t* WT2   = (bf16_t*)(ws + off); off += (size_t)128 * 512 * 2;
  float*  asb   = (float*)(ws + off);  off += (size_t)N_NODES * N_HEADS * 4;
  float*  adb   = (float*)(ws + off);  off += (size_t)N_NODES * N_HEADS * 4;
  unsigned* mb  = (unsigned*)(ws + off); off += (size_t)N_NODES * N_HEADS * 4;
  float*  den   = (float*)(ws + off);  off += (size_t)N_NODES * N_HEADS * 4;
  float*  ebuf  = (float*)(ws + off);  off += (size_t)(N_EDGES + N_NODES) * N_HEADS * 4;
  float*  psum  = (float*)(ws + off);  off += (size_t)N_GRAPH * DIM_HID * 4;
  float*  pcnt  = (float*)(ws + off);  off += (size_t)N_GRAPH * 4;

  bf16_t* WT[3]  = { WT0, WT1, WT2 };
  const int Kin[3]  = { DIM_IN, N_HEADS * DIM_HID, N_HEADS * DIM_HID };   // 128, 512, 512
  const int Fout[3] = { N_HEADS * DIM_HID, N_HEADS * DIM_HID, DIM_HID };  // 512, 512, 128
  const int Heads[3] = { N_HEADS, N_HEADS, 1 };

  const int T = 256;
  const int EN = N_EDGES + N_NODES;

  // ---- prep: transpose+convert weights, convert input features ----
  for (int li = 0; li < 3; ++li) {
    int sz = Kin[li] * Fout[li];
    k_w_transpose_bf16<<<(sz + T - 1) / T, T, 0, stream>>>(lay[li].W, WT[li], Kin[li], Fout[li]);
  }
  {
    int sz = N_NODES * DIM_IN;
    k_f32_to_bf16<<<(sz + T - 1) / T, T, 0, stream>>>(x, XB, sz);
  }

  // ---- three GAT layers ----
  for (int li = 0; li < 3; ++li) {
    const int K = Kin[li], F = Fout[li], H = Heads[li], D = F / H;

    // h = x @ W  (bf16 WMMA, fp32 accumulate)
    int tiles = (N_NODES / 16) * (F / 64);
    k_gemm_bf16_wmma<<<(tiles + 7) / 8, T, 0, stream>>>(XB, WT[li], bufA, N_NODES, K, F);

    // attention coefficients + segment-max/denom init
    int nh = N_NODES * H;
    k_attn_coef<<<(nh + T - 1) / T, T, 0, stream>>>(bufA, lay[li].as, lay[li].ad,
                                                    asb, adb, mb, den, N_NODES, H, D);
    // edge softmax
    int enh = EN * H;
    k_edge_logits<<<(enh + T - 1) / T, T, 0, stream>>>(edge_index, asb, adb, ebuf, mb, H);
    k_edge_exp<<<(enh + T - 1) / T, T, 0, stream>>>(edge_index, ebuf, mb, den, H);

    // weighted scatter aggregation
    hipMemsetAsync(bufB, 0, (size_t)N_NODES * F * sizeof(float), stream);
    long long work = (long long)EN * F;
    k_edge_aggregate<<<(unsigned)((work + T - 1) / T), T, 0, stream>>>(
        edge_index, ebuf, den, bufA, bufB, H, D);

    // +bias, BN, (ReLU), convert for next layer
    long long nf = (long long)N_NODES * F;
    if (li < 2) {
      k_bias_bn_act<<<(unsigned)((nf + T - 1) / T), T, 0, stream>>>(
          bufB, lay[li].b, lay[li].g, lay[li].bb, lay[li].m, lay[li].v,
          nullptr, XB, N_NODES, F, 1);
    } else {
      k_bias_bn_act<<<(unsigned)((nf + T - 1) / T), T, 0, stream>>>(
          bufB, lay[li].b, lay[li].g, lay[li].bb, lay[li].m, lay[li].v,
          bufA, nullptr, N_NODES, F, 0);
    }
  }

  // ---- global mean pool + classifier MLP + sigmoid ----
  hipMemsetAsync(psum, 0, (size_t)N_GRAPH * DIM_HID * sizeof(float), stream);
  hipMemsetAsync(pcnt, 0, (size_t)N_GRAPH * sizeof(float), stream);
  {
    int sz = N_NODES * DIM_HID;
    k_pool<<<(sz + T - 1) / T, T, 0, stream>>>(bufA, batch, psum, pcnt, N_NODES, DIM_HID);
  }
  k_classifier<<<1, 64, 0, stream>>>(psum, pcnt, cW[0], cB[0], cW[1], cB[1], cW[2], cB[2], out);
}
